// MultiheadCrossAttention_146028888611
// MI455X (gfx1250) — matrix operand order repaired and checked by reading, not compiled, not run on hardware
//
#include <hip/hip_runtime.h>
#include <hip/hip_bf16.h>

// ---------------------------------------------------------------------------
// CDNA5 (gfx1250) wave32 WMMA implementation of MultiheadCrossAttention.
//   kv  = x @ W_kv^T            [4,1024,2048]
//   q1  = x @ W_q1^T            [4,1024,1024]
//   q2  = x_cond @ W_q2^T       [4,1024,1024]
//   q   = W_one@q1 + W_zero@q2 + b_one + b_zero   (token mixing, per batch)
//   attn = flash-attention over qkv = [q | kv], 16 heads, dim 64
//   out = attn @ W_proj^T + b_proj                 (f32 output)
// All matmuls use v_wmma_f32_16x16x32_bf16 with f32 accumulation.
// bf16->LDS staging uses GLOBAL_LOAD_ASYNC_TO_LDS_B128 when available.
// ---------------------------------------------------------------------------

typedef __attribute__((ext_vector_type(16))) __bf16 v16bf;
typedef __attribute__((ext_vector_type(8)))  float  v8f;
typedef unsigned short ushort_t;

#define BS_   4
#define NCTX  1024
#define WIDTH 1024
#define NHEAD 16
#define HDIM  64

// Native f32 -> bf16 (RNE); lets ISel use v_cvt_*bf16* instead of int ops.
static __device__ __forceinline__ ushort_t f2bf(float f) {
    union { __bf16 b; ushort_t u; } c;
    c.b = (__bf16)f;
    return c.u;
}

static __device__ __forceinline__ v8f v8f_zero() {
    v8f z;
#pragma unroll
    for (int i = 0; i < 8; ++i) z[i] = 0.0f;
    return z;
}

// A-fragment (16x32 bf16), lane = 16h + m.  elems 0..7 : k = kBase + 8h + e
//                                           elems 8..15: k = kBase + 16 + 8h + (e-8)
static __device__ __forceinline__ v16bf load_a_frag(const ushort_t* lds_row, int h, int kBase) {
    union { v16bf v; uint4 q[2]; } r;
    r.q[0] = *(const uint4*)(lds_row + kBase + 8 * h);
    r.q[1] = *(const uint4*)(lds_row + kBase + 16 + 8 * h);
    return r.v;
}

// B-fragment (32x16 bf16), lane = 16h + n.  elems 0..7 : k = kBase + 8h + e
//                                           elems 8..15: k = kBase + 16 + 8h + (e-8)
static __device__ __forceinline__ v16bf load_b_frag(const ushort_t* lds_row, int h, int kBase) {
    union { v16bf v; uint4 q[2]; } r;
    r.q[0] = *(const uint4*)(lds_row + kBase + 8 * h);
    r.q[1] = *(const uint4*)(lds_row + kBase + 16 + 8 * h);
    return r.v;
}

static __device__ __forceinline__ v8f wmma_bf16(v16bf a, v16bf b, v8f c) {
    return __builtin_amdgcn_wmma_f32_16x16x32_bf16(false, a, false, b, (short)0, c, false, false);
}

// ---------------------------------------------------------------------------
// Async global->LDS (gfx1250 GLOBAL_LOAD_ASYNC_TO_LDS_B128, ASYNCcnt-tracked).
// The builtin's parameters are non-const pointers to 4xi32 vectors in
// addrspace(1)/(3) (revealed by the compiler diagnostic).  Generic pointers
// are converted via uintptr_t: low 32 bits of a generic LDS address are the
// LDS byte offset (ISA 00_overview aperture rules).
// ---------------------------------------------------------------------------
typedef int v4i_vec __attribute__((vector_size(4 * sizeof(int))));
typedef __attribute__((address_space(1))) v4i_vec* gptr_v4i;
typedef __attribute__((address_space(3))) v4i_vec* lptr_v4i;

#if __has_builtin(__builtin_amdgcn_global_load_async_to_lds_b128)
#define HAVE_ASYNC_LDS 1
static __device__ __forceinline__ void copy16_async(ushort_t* lds_dst, const ushort_t* gsrc) {
    __builtin_amdgcn_global_load_async_to_lds_b128(
        (gptr_v4i)(uintptr_t)gsrc,
        (lptr_v4i)(uintptr_t)lds_dst,
        0, 0);
}
#else
#define HAVE_ASYNC_LDS 0
static __device__ __forceinline__ void copy16_async(ushort_t* lds_dst, const ushort_t* gsrc) {
    *(uint4*)lds_dst = *(const uint4*)gsrc;
}
#endif

static __device__ __forceinline__ void wait_async() {
#if HAVE_ASYNC_LDS
#if __has_builtin(__builtin_amdgcn_s_wait_asynccnt)
    __builtin_amdgcn_s_wait_asynccnt(0);
#else
    asm volatile("s_wait_asynccnt 0x0" ::: "memory");
#endif
#endif
}

// Stage 8 contiguous elements into LDS, converting f32 -> bf16.
static __device__ __forceinline__ void stage8(ushort_t* dst, const float* src) {
    const float4 a = *(const float4*)src;
    const float4 b = *(const float4*)(src + 4);
    dst[0] = f2bf(a.x); dst[1] = f2bf(a.y); dst[2] = f2bf(a.z); dst[3] = f2bf(a.w);
    dst[4] = f2bf(b.x); dst[5] = f2bf(b.y); dst[6] = f2bf(b.z); dst[7] = f2bf(b.w);
}
static __device__ __forceinline__ void stage8(ushort_t* dst, const ushort_t* src) {
    copy16_async(dst, src);                   // bf16 source: raw async copy
}
static __device__ __forceinline__ void maybe_wait(const float*)    {}
static __device__ __forceinline__ void maybe_wait(const ushort_t*) { wait_async(); }

static __device__ __forceinline__ void store_out(float* p, float v)    { *p = v; }
static __device__ __forceinline__ void store_out(ushort_t* p, float v) { *p = f2bf(v); }

// ---------------------------------------------------------------------------
// Kernel 1: C[M,N] = A[M,K] @ W[N,K]^T + bias[N].  M,N % 128 == 0, K % 32 == 0.
// 256 threads = 8 waves (2x4).  Each wave: 64x32 output (4x2 WMMA tiles).
// ---------------------------------------------------------------------------
template <typename TA, typename TO>
__global__ __launch_bounds__(256)
void gemm_bias_kernel(const TA* __restrict__ A, const float* __restrict__ W,
                      const float* __restrict__ bias, TO* __restrict__ C,
                      int M, int N, int K)
{
    constexpr int BK = 32, LDT = BK + 8;      // padded LDS row stride (shorts)
    __shared__ ushort_t As[128 * LDT];
    __shared__ ushort_t Ws[128 * LDT];

    const int tid  = threadIdx.x;
    const int lane = tid & 31, wid = tid >> 5;
    const int h = lane >> 4, ln = lane & 15;
    const int wm = wid >> 2, wn = wid & 3;    // wave grid 2(M) x 4(N)
    const int bm = blockIdx.y * 128, bn = blockIdx.x * 128;

    v8f acc[4][2];
#pragma unroll
    for (int i = 0; i < 4; ++i)
#pragma unroll
        for (int j = 0; j < 2; ++j) acc[i][j] = v8f_zero();

    const int srow = tid >> 2, scol = (tid & 3) * 8;   // staging: 8 elems/thread/pass

    for (int k0 = 0; k0 < K; k0 += BK) {
#pragma unroll
        for (int p = 0; p < 2; ++p) {
            const int r = srow + 64 * p;
            stage8(As + r * LDT + scol, A + (size_t)(bm + r) * K + k0 + scol);
            stage8(Ws + r * LDT + scol, W + (size_t)(bn + r) * K + k0 + scol);
        }
        maybe_wait(A);
        __syncthreads();

        v16bf bfr[2];
#pragma unroll
        for (int j = 0; j < 2; ++j)
            bfr[j] = load_b_frag(Ws + (wn * 32 + j * 16 + ln) * LDT, h, 0);
#pragma unroll
        for (int i = 0; i < 4; ++i) {
            const v16bf afr = load_a_frag(As + (wm * 64 + i * 16 + ln) * LDT, h, 0);
#pragma unroll
            for (int j = 0; j < 2; ++j)
                acc[i][j] = wmma_bf16(afr, bfr[j], acc[i][j]);
        }
        __syncthreads();
    }

#pragma unroll
    for (int i = 0; i < 4; ++i)
#pragma unroll
        for (int j = 0; j < 2; ++j) {
            const int n = bn + wn * 32 + j * 16 + ln;
            const float bv = bias[n];
#pragma unroll
            for (int r = 0; r < 8; ++r) {
                const int m = bm + wm * 64 + i * 16 + r + 8 * h;
                store_out(C + (size_t)m * N + n, acc[i][j][r] + bv);
            }
        }
}

// ---------------------------------------------------------------------------
// Kernel 2: token mixing, per batch (grid.z):
//   Qout[t,w] = sum_s W1[t,s]*Q1[b,s,w] + W0[t,s]*Q2[b,s,w] + b1[t] + b0[t]
// B operands staged transposed ([w][s]) so B-fragment reads are contiguous.
// ---------------------------------------------------------------------------
__global__ __launch_bounds__(256)
void gemm_mix_kernel(const float* __restrict__ W1, const float* __restrict__ W0,
                     const ushort_t* __restrict__ Q1, const ushort_t* __restrict__ Q2,
                     const float* __restrict__ b1, const float* __restrict__ b0,
                     ushort_t* __restrict__ Qout)
{
    constexpr int K = NCTX, Wd = WIDTH, LDT = 40;
    __shared__ ushort_t A1s[128 * LDT];
    __shared__ ushort_t A0s[128 * LDT];
    __shared__ ushort_t B1t[128 * LDT];
    __shared__ ushort_t B0t[128 * LDT];

    const int tid  = threadIdx.x;
    const int lane = tid & 31, wid = tid >> 5;
    const int h = lane >> 4, ln = lane & 15;
    const int wm = wid >> 2, wn = wid & 3;
    const int bt = blockIdx.y * 128, bw = blockIdx.x * 128;
    const int b  = blockIdx.z;

    const ushort_t* Q1b = Q1 + (size_t)b * NCTX * Wd;
    const ushort_t* Q2b = Q2 + (size_t)b * NCTX * Wd;
    ushort_t*       Qob = Qout + (size_t)b * NCTX * Wd;

    v8f acc[4][2];
#pragma unroll
    for (int i = 0; i < 4; ++i)
#pragma unroll
        for (int j = 0; j < 2; ++j) acc[i][j] = v8f_zero();

    const int arow = tid >> 2, acol = (tid & 3) * 8;    // A staging
    const int brow = tid >> 4, bcol = (tid & 15) * 8;   // B staging (transpose)

    for (int s0 = 0; s0 < K; s0 += 32) {
#pragma unroll
        for (int p = 0; p < 2; ++p) {
            const int r = arow + 64 * p;
            stage8(A1s + r * LDT + acol, W1 + (size_t)(bt + r) * K + s0 + acol);
            stage8(A0s + r * LDT + acol, W0 + (size_t)(bt + r) * K + s0 + acol);
        }
#pragma unroll
        for (int p = 0; p < 2; ++p) {
            const int sr = brow + 16 * p;               // 0..31 within K-tile
            union { uint4 q; ushort_t u[8]; } v1, v0;
            v1.q = *(const uint4*)(Q1b + (size_t)(s0 + sr) * Wd + bw + bcol);
            v0.q = *(const uint4*)(Q2b + (size_t)(s0 + sr) * Wd + bw + bcol);
#pragma unroll
            for (int e = 0; e < 8; ++e) {
                B1t[(bcol + e) * LDT + sr] = v1.u[e];
                B0t[(bcol + e) * LDT + sr] = v0.u[e];
            }
        }
        __syncthreads();

        v16bf b1f[2], b0f[2];
#pragma unroll
        for (int j = 0; j < 2; ++j) {
            b1f[j] = load_b_frag(B1t + (wn * 32 + j * 16 + ln) * LDT, h, 0);
            b0f[j] = load_b_frag(B0t + (wn * 32 + j * 16 + ln) * LDT, h, 0);
        }
#pragma unroll
        for (int i = 0; i < 4; ++i) {
            const v16bf a1f = load_a_frag(A1s + (wm * 64 + i * 16 + ln) * LDT, h, 0);
            const v16bf a0f = load_a_frag(A0s + (wm * 64 + i * 16 + ln) * LDT, h, 0);
#pragma unroll
            for (int j = 0; j < 2; ++j) {
                acc[i][j] = wmma_bf16(a1f, b1f[j], acc[i][j]);
                acc[i][j] = wmma_bf16(a0f, b0f[j], acc[i][j]);
            }
        }
        __syncthreads();
    }

#pragma unroll
    for (int i = 0; i < 4; ++i)
#pragma unroll
        for (int j = 0; j < 2; ++j) {
            const int w = bw + wn * 32 + j * 16 + ln;
#pragma unroll
            for (int r = 0; r < 8; ++r) {
                const int t = bt + wm * 64 + i * 16 + r + 8 * h;
                const float bv = b1[t] + b0[t];
                Qob[(size_t)t * Wd + w] = f2bf(acc[i][j][r] + bv);
            }
        }
}

// ---------------------------------------------------------------------------
// Kernel 3: flash attention.  qkv channel c of head hd: role r -> c0 = hd*192+r*64;
// channels [c0, c0+64) come entirely from q_buf (c0 < 1024) or kv_buf (else),
// since 1024 is a multiple of 64.  grid = (NCTX/128, NHEAD, BS).
// Each of 8 waves owns 16 query rows; keys streamed in blocks of 64.
// Q and K tiles staged with async global->LDS copies; V transposed manually.
// ---------------------------------------------------------------------------
__global__ __launch_bounds__(256)
void flash_attn_kernel(const ushort_t* __restrict__ Qbuf,   // [B,NCTX,1024]
                       const ushort_t* __restrict__ KVbuf,  // [B,NCTX,2048]
                       ushort_t* __restrict__ Obuf)         // [B,NCTX,1024]
{
    constexpr int BKEY = 64, LQ = HDIM + 8, LK = HDIM + 8, LV = BKEY + 8, LP = BKEY + 8;
    __shared__ ushort_t Qs[128 * LQ];
    __shared__ ushort_t Ks[BKEY * LK];
    __shared__ ushort_t Vt[HDIM * LV];          // transposed: [d][key]
    __shared__ ushort_t Ps[8 * 16 * LP];        // per-wave P tiles

    const int tid  = threadIdx.x;
    const int lane = tid & 31, wid = tid >> 5;
    const int h = lane >> 4, ln = lane & 15;
    const int mblk = blockIdx.x, hd = blockIdx.y, b = blockIdx.z;

    // role pointers + row strides into the virtual qkv tensor
    const ushort_t* base[3]; int strd[3];
#pragma unroll
    for (int role = 0; role < 3; ++role) {
        const int c0 = hd * 192 + role * 64;
        if (c0 < WIDTH) { base[role] = Qbuf  + (size_t)b * NCTX * WIDTH     + c0;           strd[role] = WIDTH; }
        else            { base[role] = KVbuf + (size_t)b * NCTX * (2*WIDTH) + (c0 - WIDTH); strd[role] = 2 * WIDTH; }
    }

    // load Q block [128][64] (async)
    {
        const int row = tid >> 3, col = (tid & 7) * 8;
#pragma unroll
        for (int p = 0; p < 4; ++p) {
            const int r = row + 32 * p;
            copy16_async(Qs + r * LQ + col,
                         base[0] + (size_t)(mblk * 128 + r) * strd[0] + col);
        }
    }
    wait_async();
    __syncthreads();

    const ushort_t* qrow = Qs + (wid * 16 + ln) * LQ;
    const v16bf aq0 = load_a_frag(qrow, h, 0);
    const v16bf aq1 = load_a_frag(qrow, h, 32);

    v8f oacc[4];
#pragma unroll
    for (int j = 0; j < 4; ++j) oacc[j] = v8f_zero();
    float mrow[8], lrow[8];
#pragma unroll
    for (int r = 0; r < 8; ++r) { mrow[r] = -3.0e38f; lrow[r] = 0.0f; }

    const float SCALE = 0.125f * 1.44269504088896340736f;   // 64^-1/2 * log2(e)
    ushort_t* myP = Ps + wid * 16 * LP;

    for (int kb = 0; kb < NCTX / BKEY; ++kb) {
        // stage K [64][64] (async) and V transposed [64 d][64 key] (manual)
        {
            const int row = tid >> 3, col = (tid & 7) * 8;
#pragma unroll
            for (int p = 0; p < 2; ++p) {
                const int r = row + 32 * p;
                copy16_async(Ks + r * LK + col,
                             base[1] + (size_t)(kb * BKEY + r) * strd[1] + col);
                union { uint4 q; ushort_t u[8]; } vv;
                vv.q = *(const uint4*)(base[2] + (size_t)(kb * BKEY + r) * strd[2] + col);
#pragma unroll
                for (int e = 0; e < 8; ++e) Vt[(col + e) * LV + r] = vv.u[e];
            }
        }
        wait_async();
        __syncthreads();

        // S = Q @ K^T  (16 query rows x 64 keys per wave)
        v8f s8[4];
#pragma unroll
        for (int nt = 0; nt < 4; ++nt) {
            s8[nt] = v8f_zero();
            const ushort_t* krow = Ks + (nt * 16 + ln) * LK;
            s8[nt] = wmma_bf16(aq0, load_b_frag(krow, h, 0),  s8[nt]);
            s8[nt] = wmma_bf16(aq1, load_b_frag(krow, h, 32), s8[nt]);
        }

        // online softmax (exp2 domain)
#pragma unroll
        for (int r = 0; r < 8; ++r) {
            float mx = -3.0e38f;
#pragma unroll
            for (int nt = 0; nt < 4; ++nt) mx = fmaxf(mx, s8[nt][r]);
#pragma unroll
            for (int off = 1; off < 16; off <<= 1)
                mx = fmaxf(mx, __shfl_xor(mx, off, 32));
            const float mnew = fmaxf(mrow[r], mx * SCALE);
            const float corr = exp2f(mrow[r] - mnew);
            mrow[r] = mnew;
            float rsum = 0.0f;
#pragma unroll
            for (int nt = 0; nt < 4; ++nt) {
                const float pv = exp2f(s8[nt][r] * SCALE - mnew);
                s8[nt][r] = pv;
                rsum += pv;
            }
#pragma unroll
            for (int off = 1; off < 16; off <<= 1)
                rsum += __shfl_xor(rsum, off, 32);
            lrow[r] = lrow[r] * corr + rsum;
#pragma unroll
            for (int j = 0; j < 4; ++j) oacc[j][r] *= corr;
        }

        // P -> LDS (accumulator layout -> A-fragment layout)
#pragma unroll
        for (int nt = 0; nt < 4; ++nt)
#pragma unroll
            for (int r = 0; r < 8; ++r)
                myP[(r + 8 * h) * LP + nt * 16 + ln] = f2bf(s8[nt][r]);
        __syncthreads();

        // O += P @ V
#pragma unroll
        for (int ks = 0; ks < 2; ++ks) {
            const v16bf ap = load_a_frag(myP + ln * LP, h, ks * 32);
#pragma unroll
            for (int j = 0; j < 4; ++j) {
                const v16bf bv = load_b_frag(Vt + (j * 16 + ln) * LV, h, ks * 32);
                oacc[j] = wmma_bf16(ap, bv, oacc[j]);
            }
        }
        __syncthreads();
    }

    // epilogue: O / l -> Obuf[b, t, hd*64 + d]
#pragma unroll
    for (int j = 0; j < 4; ++j) {
        const int d = j * 16 + ln;
#pragma unroll
        for (int r = 0; r < 8; ++r) {
            const int t = mblk * 128 + wid * 16 + r + 8 * h;
            const float v = oacc[j][r] / lrow[r];
            Obuf[((size_t)b * NCTX + t) * WIDTH + hd * HDIM + d] = f2bf(v);
        }
    }
}

// ---------------------------------------------------------------------------
// Host-side launcher
// ---------------------------------------------------------------------------
extern "C" void kernel_launch(void* const* d_in, const int* in_sizes, int n_in,
                              void* d_out, int out_size, void* d_ws, size_t ws_size,
                              hipStream_t stream)
{
    const float* x      = (const float*)d_in[0];
    const float* x_cond = (const float*)d_in[1];
    const float* W_kv   = (const float*)d_in[2];
    const float* b_kv   = (const float*)d_in[3];
    const float* W_q1   = (const float*)d_in[4];
    const float* b_q1   = (const float*)d_in[5];
    const float* W_q2   = (const float*)d_in[6];
    const float* b_q2   = (const float*)d_in[7];
    const float* W_proj = (const float*)d_in[8];
    const float* b_proj = (const float*)d_in[9];
    const float* W_one  = (const float*)d_in[10];
    const float* b_one  = (const float*)d_in[11];
    const float* W_zero = (const float*)d_in[12];
    const float* b_zero = (const float*)d_in[13];
    float* out = (float*)d_out;

    const int M = BS_ * NCTX;                       // 4096
    char* ws = (char*)d_ws;
    ushort_t* kv_buf   = (ushort_t*)(ws);                                        // [4096,2048]
    ushort_t* q1_buf   = (ushort_t*)(ws + (size_t)M * 2 * WIDTH * 2);            // [4096,1024]
    ushort_t* q2_buf   = (ushort_t*)(ws + (size_t)M * 3 * WIDTH * 2);
    ushort_t* q_buf    = (ushort_t*)(ws + (size_t)M * 4 * WIDTH * 2);
    ushort_t* attn_buf = (ushort_t*)(ws + (size_t)M * 5 * WIDTH * 2);

    dim3 blk(256);
    // 1) input projections
    gemm_bias_kernel<float, ushort_t><<<dim3((2 * WIDTH) / 128, M / 128), blk, 0, stream>>>(
        x, W_kv, b_kv, kv_buf, M, 2 * WIDTH, WIDTH);
    gemm_bias_kernel<float, ushort_t><<<dim3(WIDTH / 128, M / 128), blk, 0, stream>>>(
        x, W_q1, b_q1, q1_buf, M, WIDTH, WIDTH);
    gemm_bias_kernel<float, ushort_t><<<dim3(WIDTH / 128, M / 128), blk, 0, stream>>>(
        x_cond, W_q2, b_q2, q2_buf, M, WIDTH, WIDTH);
    // 2) token mixing (1x1 conv over token axis)
    gemm_mix_kernel<<<dim3(WIDTH / 128, NCTX / 128, BS_), blk, 0, stream>>>(
        W_one, W_zero, q1_buf, q2_buf, b_one, b_zero, q_buf);
    // 3) flash attention
    flash_attn_kernel<<<dim3(NCTX / 128, NHEAD, BS_), blk, 0, stream>>>(
        q_buf, kv_buf, attn_buf);
    // 4) output projection (f32 out)
    gemm_bias_kernel<ushort_t, float><<<dim3(WIDTH / 128, M / 128), blk, 0, stream>>>(
        attn_buf, W_proj, b_proj, out, M, WIDTH, WIDTH);
}
